// Decoder_40046275068363
// MI455X (gfx1250) — matrix operand order repaired and checked by reading, not compiled, not run on hardware
//
#include <hip/hip_runtime.h>
#include <hip/hip_bf16.h>
#include <math.h>

// ---------- CDNA5 WMMA types ----------
typedef __bf16 bf16;
typedef __attribute__((ext_vector_type(16))) __bf16 v16bf;
typedef __attribute__((ext_vector_type(8)))  float  v8f;

// Dims pinned by the module
#define T_STEPS 512
#define B       128
#define HE      512
#define DEMB    128
#define DHID    512
#define VOC     33
#define KCAT    640          // DEMB + DHID
#define G4      2048         // 4*DHID

// ---------------------------------------------------------------------------
// Fragment layout for V_WMMA_F32_16X16X32_BF16 (wave32).
// A (16x32, MxK): lane L holds row m=L&15; elems [kb..kb+7] and [kb+16..kb+23],
//                 kb=(L>>4)*8  -> two contiguous 16-byte reads.
// B (32x16, KxN): lane L holds col n=L&15; elems [kb..kb+7] and [kb+16..kb+23], kb=(L>>4)*8,
//                 sourced K-contiguous (row-major N x K) -> two contiguous 16-byte reads.
// C/D (16x16):   lane L holds col n=L&15, rows (L>>4)*8 + r, r=0..7.
// ---------------------------------------------------------------------------
__device__ inline v16bf load_a_frag(const bf16* __restrict__ Arow, int kbA) {
  v16bf a;
#pragma unroll
  for (int e = 0; e < 8; ++e) a[e] = Arow[kbA + e];
#pragma unroll
  for (int e = 0; e < 8; ++e) a[8 + e] = Arow[kbA + 16 + e];
  return a;
}

// ---------------------------------------------------------------------------
// Unified WMMA GEMM:  C(M x N) = A(M x K) * Bt(N x K)^T  [+ bias0 + bias1]
// A, Bt row-major bf16, K-contiguous.  Block = 256 thr = 8 waves.
// Wave w -> M-tile (blockIdx.y*128 + w*16); block covers 64 N columns.
// B tile (32K x 64N = 4KB) shared by all 8 waves -> LDS, double-buffered.
// Main loop is straight-line (last K-step peeled): stage next tile,
// near-scope global_prefetch two tiles ahead, register-prefetch next A
// fragment, and a 2-deep B-fragment pipeline so each dscnt wait overlaps
// the previous WMMA's execution.
// ---------------------------------------------------------------------------
template <int KDIM, bool HAS_BIAS>
__global__ __launch_bounds__(256)
void wmma_gemm_kernel(const bf16* __restrict__ A, int lda,
                      const bf16* __restrict__ Bt, int ldb,
                      const float* __restrict__ bias0,
                      const float* __restrict__ bias1,
                      float* __restrict__ C, int ldc) {
  constexpr int NITER = KDIM / 32;
  __shared__ bf16 Bs[2][64 * 32];          // [buf][n_local*32 + k]

  const int tid   = threadIdx.x;
  const int lane  = tid & 31;
  const int wave  = tid >> 5;
  const int mBase = blockIdx.y * 128 + wave * 16;
  const int nBase = blockIdx.x * 64;
  const int kbA   = (lane >> 4) * 8;
  const int kbB   = (lane >> 4) * 16;
  const int m     = mBase + (lane & 15);
  const bf16* __restrict__ Arow = A + (size_t)m * lda;

  // Cooperative B staging: each thread moves 8 bf16 (16 bytes).
  const int srow = tid >> 2;               // 0..63 (n_local)
  const int scol = (tid & 3) * 8;          // 0,8,16,24 (k)
  const bf16* __restrict__ Bsrc = Bt + (size_t)(nBase + srow) * ldb + scol;

  auto stage = [&](int buf, int k0) {
#pragma unroll
    for (int e = 0; e < 8; ++e)
      Bs[buf][srow * 32 + scol + e] = Bsrc[k0 + e];
  };

  auto load_b = [&](const bf16* __restrict__ BsCur, int nn) {
    v16bf b;
#pragma unroll
    for (int e = 0; e < 16; ++e)
      b[e] = BsCur[(nn * 16 + (lane & 15)) * 32 + kbA + (e & 7) + ((e >> 3) << 4)];
    return b;
  };

  // 2-deep pipelined 4-WMMA burst: issue loads for nn+1 before WMMA on nn.
  auto mma4 = [&](const bf16* __restrict__ BsCur, const v16bf& a, v8f (&acc)[4]) {
    v16bf bcur = load_b(BsCur, 0);
#pragma unroll
    for (int nn = 0; nn < 4; ++nn) {
      v16bf bnext = bcur;
      if (nn + 1 < 4) bnext = load_b(BsCur, nn + 1);
      acc[nn] = __builtin_amdgcn_wmma_f32_16x16x32_bf16(
          false, a, false, bcur, (short)0, acc[nn], false, false);
      bcur = bnext;
    }
  };

  stage(0, 0);
  v16bf a = load_a_frag(Arow, kbA);        // K-step 0
  v8f acc[4] = {};

#pragma unroll 1
  for (int it = 0; it < NITER - 1; ++it) {
    __syncthreads();                       // Bs[it&1] ready for all waves
    const int k0 = it * 32;
    stage((it + 1) & 1, k0 + 32);          // next tile -> other LDS buffer
    __builtin_prefetch(Bsrc + k0 + 64, 0, 3);   // near-scope, 2 tiles ahead
    __builtin_prefetch(Arow + k0 + 64, 0, 3);
    v16bf aNext = load_a_frag(Arow + k0 + 32, kbA);
    mma4(Bs[it & 1], a, acc);
    a = aNext;
  }

  // Epilogue: final K-step (no staging / no A prefetch)
  __syncthreads();
  mma4(Bs[(NITER - 1) & 1], a, acc);

  const int rowOff = (lane >> 4) * 8;
#pragma unroll
  for (int nn = 0; nn < 4; ++nn) {
    const int n = nBase + nn * 16 + (lane & 15);
    float bias = 0.f;
    if constexpr (HAS_BIAS) bias = bias0[n] + bias1[n];
#pragma unroll
    for (int r = 0; r < 8; ++r)
      C[(size_t)(mBase + rowOff + r) * ldc + n] = acc[nn][r] + bias;
  }
}

// ---------------------------------------------------------------------------
// Setup: Wcat[n][k] = (k<128) ? Wi[n][k] : Wh[n][k-128]   (bf16, 2048x640)
// ---------------------------------------------------------------------------
__global__ void build_wcat_kernel(const float* __restrict__ Wi,
                                  const float* __restrict__ Wh,
                                  bf16* __restrict__ Wcat) {
  int idx = blockIdx.x * blockDim.x + threadIdx.x;
  if (idx >= G4 * KCAT) return;
  int n = idx / KCAT, k = idx - n * KCAT;
  float v = (k < DEMB) ? Wi[n * DEMB + k] : Wh[n * DHID + (k - DEMB)];
  Wcat[idx] = (bf16)v;
}

// ---------------------------------------------------------------------------
// Step 1: emb = y @ We^T + be  -> bf16 into X[:, 0:128]
// ---------------------------------------------------------------------------
__global__ void embed_kernel(const float* __restrict__ y,
                             const float* __restrict__ We,
                             const float* __restrict__ be,
                             bf16* __restrict__ X) {
  int idx = blockIdx.x * blockDim.x + threadIdx.x;   // B*DEMB = 16384
  int b = idx >> 7, d = idx & 127;
  float s = be[d];
#pragma unroll
  for (int v = 0; v < VOC; ++v) s += y[b * VOC + v] * We[d * VOC + v];
  X[b * KCAT + d] = (bf16)s;
}

// ---------------------------------------------------------------------------
// Step 2: heT[j][b] = enc_h[t][b][j]   (bf16, 512x128, K-contiguous WMMA A)
// ---------------------------------------------------------------------------
__global__ void henc_transpose_kernel(const float* __restrict__ he,
                                      bf16* __restrict__ heT) {
  int idx = blockIdx.x * blockDim.x + threadIdx.x;   // HE*B = 65536
  int j = idx >> 7, b = idx & 127;
  heT[idx] = (bf16)he[b * HE + j];
}

// ---------------------------------------------------------------------------
// Step 4: LSTM pointwise.  torch gate order i,f,g,o.
// Writes h (f32), X[:,128+n] (bf16, next-step GEMM A), hT[n][b] (bf16 score B).
// ---------------------------------------------------------------------------
__device__ inline float sigmoidf(float x) { return 1.0f / (1.0f + expf(-x)); }

__global__ void lstm_kernel(const float* __restrict__ gates,
                            float* __restrict__ c,
                            float* __restrict__ h,
                            bf16* __restrict__ X,
                            bf16* __restrict__ hT) {
  int idx = blockIdx.x * blockDim.x + threadIdx.x;   // B*DHID = 65536
  int b = idx >> 9, n = idx & 511;
  const float* g = gates + b * G4;
  float ig = sigmoidf(g[n]);
  float fg = sigmoidf(g[DHID + n]);
  float gg = tanhf(g[2 * DHID + n]);
  float og = sigmoidf(g[3 * DHID + n]);
  float cn = fg * c[idx] + ig * gg;
  float hn = og * tanhf(cn);
  c[idx] = cn;
  h[idx] = hn;
  X[b * KCAT + DEMB + n] = (bf16)hn;
  hT[n * B + b] = (bf16)hn;
}

// ---------------------------------------------------------------------------
// Step 6: arow[n] = (sum_j exp(score[j][n])) / (sum_j score[j][n])
// (matches the reference's non-softmax: denom is sum of RAW scores)
// ---------------------------------------------------------------------------
__global__ void attn_reduce_kernel(const float* __restrict__ score,
                                   float* __restrict__ arow) {
  int n = blockIdx.x * blockDim.x + threadIdx.x;     // 512
  float s1 = 0.f, s2 = 0.f;
  for (int j = 0; j < HE; ++j) {
    float s = score[j * HE + n];
    s1 += s;
    s2 += expf(s);
  }
  arow[n] = s2 / s1;
}

// ---------------------------------------------------------------------------
// Step 7: ct += arow * h_e; y = [h|ct]@Wo^T + bo; preds = log_softmax(y)
// One block per batch row.
// ---------------------------------------------------------------------------
__global__ __launch_bounds__(256)
void out_kernel(const float* __restrict__ he,
                const float* __restrict__ h,
                float* __restrict__ ct,
                const float* __restrict__ arow,
                const float* __restrict__ Wo,
                const float* __restrict__ bo,
                float* __restrict__ y,
                float* __restrict__ preds) {
  int b = blockIdx.x, tid = threadIdx.x;
  __shared__ float cat[DHID + HE];   // [h | ct] row
  __shared__ float yv[VOC];
  __shared__ float lse;

  for (int j = tid; j < HE; j += 256) {
    float ctv = ct[b * HE + j] + arow[j] * he[b * HE + j];
    ct[b * HE + j] = ctv;
    cat[DHID + j] = ctv;
    cat[j] = h[b * DHID + j];
  }
  __syncthreads();

  int wave = tid >> 5, lane = tid & 31;
  for (int v = wave; v < VOC; v += 8) {
    float p = 0.f;
    const float* w = Wo + v * (DHID + HE);
    for (int k = lane; k < DHID + HE; k += 32) p += cat[k] * w[k];
#pragma unroll
    for (int off = 16; off > 0; off >>= 1) p += __shfl_xor(p, off, 32);
    if (lane == 0) yv[v] = p + bo[v];
  }
  __syncthreads();

  if (tid == 0) {
    float m = yv[0];
    for (int v = 1; v < VOC; ++v) m = fmaxf(m, yv[v]);
    float s = 0.f;
    for (int v = 0; v < VOC; ++v) s += expf(yv[v] - m);
    lse = m + logf(s);
  }
  __syncthreads();

  for (int v = tid; v < VOC; v += 256) {
    float val = yv[v];
    y[b * VOC + v] = val;          // feeds next step's embed
    preds[b * VOC + v] = val - lse;
  }
}

// ---------------------------------------------------------------------------
// Host launcher
// ---------------------------------------------------------------------------
extern "C" void kernel_launch(void* const* d_in, const int* in_sizes, int n_in,
                              void* d_out, int out_size, void* d_ws, size_t ws_size,
                              hipStream_t stream) {
  (void)in_sizes; (void)n_in; (void)out_size; (void)ws_size;
  const float* enc_h = (const float*)d_in[0];
  const float* We    = (const float*)d_in[1];
  const float* be    = (const float*)d_in[2];
  const float* Wi    = (const float*)d_in[3];
  const float* Wh    = (const float*)d_in[4];
  const float* bi    = (const float*)d_in[5];
  const float* bh    = (const float*)d_in[6];
  const float* Wo    = (const float*)d_in[7];
  const float* bo    = (const float*)d_in[8];
  float* preds = (float*)d_out;

  // Workspace carve-up (256-byte aligned slabs), ~6 MB total -> L2-resident
  char*  base = (char*)d_ws;
  size_t off  = 0;
  auto take = [&](size_t bytes) -> char* {
    char* p = base + off;
    off += (bytes + 255) & ~(size_t)255;
    return p;
  };
  bf16*  Wcat  = (bf16*)take((size_t)G4 * KCAT * sizeof(bf16));   // 2.62 MB
  bf16*  X     = (bf16*)take((size_t)B * KCAT * sizeof(bf16));    // [emb|h] bf16
  bf16*  heT   = (bf16*)take((size_t)HE * B * sizeof(bf16));
  bf16*  hT    = (bf16*)take((size_t)DHID * B * sizeof(bf16));
  float* gates = (float*)take((size_t)B * G4 * sizeof(float));    // 1 MB
  float* score = (float*)take((size_t)HE * HE * sizeof(float));   // 1 MB
  float* h     = (float*)take((size_t)B * DHID * sizeof(float));
  float* c     = (float*)take((size_t)B * DHID * sizeof(float));
  float* ct    = (float*)take((size_t)B * HE * sizeof(float));
  float* y     = (float*)take((size_t)B * VOC * sizeof(float));
  float* arow  = (float*)take((size_t)HE * sizeof(float));

  // Zero the recurrent state every call (deterministic; graph-capture safe)
  hipMemsetAsync(y,  0, (size_t)B * VOC * sizeof(float), stream);
  hipMemsetAsync(h,  0, (size_t)B * DHID * sizeof(float), stream);
  hipMemsetAsync(c,  0, (size_t)B * DHID * sizeof(float), stream);
  hipMemsetAsync(ct, 0, (size_t)B * HE * sizeof(float), stream);
  hipMemsetAsync(X,  0, (size_t)B * KCAT * sizeof(bf16), stream);

  // One-time weight pack: Wcat = [Wi | Wh] in bf16, row-major (N x K)
  {
    int n = G4 * KCAT;
    build_wcat_kernel<<<(n + 255) / 256, 256, 0, stream>>>(Wi, Wh, Wcat);
  }

  for (int t = 0; t < T_STEPS; ++t) {
    const float* he_t = enc_h + (size_t)t * B * HE;
    embed_kernel        <<<(B * DEMB) / 256, 256, 0, stream>>>(y, We, be, X);
    henc_transpose_kernel<<<(HE * B) / 256, 256, 0, stream>>>(he_t, heT);
    // gates(128x2048) = X(128x640) @ Wcat^T + bi + bh
    wmma_gemm_kernel<KCAT, true><<<dim3(G4 / 64, 1), 256, 0, stream>>>(
        X, KCAT, Wcat, KCAT, bi, bh, gates, G4);
    lstm_kernel         <<<(B * DHID) / 256, 256, 0, stream>>>(gates, c, h, X, hT);
    // score(512x512) = heT(512x128) @ hT^T
    wmma_gemm_kernel<B, false><<<dim3(HE / 64, HE / 128), 256, 0, stream>>>(
        heT, B, hT, B, nullptr, nullptr, score, HE);
    attn_reduce_kernel  <<<HE / 256, 256, 0, stream>>>(score, arow);
    out_kernel          <<<B, 256, 0, stream>>>(he_t, h, ct, arow, Wo, bo, y,
                                                preds + (size_t)t * B * VOC);
  }
}
